// AddictiveAttention_80642305949871
// MI455X (gfx1250) — matrix operand order repaired and checked by reading, compiled, not run on hardware
//
#include <hip/hip_runtime.h>
#include <hip/hip_bf16.h>

// Problem constants (match reference)
#define BATCH 32
#define CDIM  512
#define HW    1024
#define HID   256
#define MEMD  256

typedef __attribute__((ext_vector_type(16))) __bf16 v16bf;
typedef __attribute__((ext_vector_type(8)))  float  v8f;

// CDNA5 async global->LDS copy (ASYNCcnt-tracked). Inline asm: portable across
// ROCm7.2 / amdgpu-toolchain (builtin arity differs between them).
#define ASYNC_LOAD_B32(laddr_u32, gaddr) \
    asm volatile("global_load_async_to_lds_b32 %0, %1, off" \
                 :: "v"(laddr_u32), "v"(gaddr) : "memory")
#define WAIT_ASYNCCNT0() asm volatile("s_wait_asynccnt 0x0" ::: "memory")

static __device__ inline unsigned int lds_off(const void* p) {
    // generic pointer to __shared__: low 32 bits == wave-relative LDS byte offset
    return (unsigned int)(uintptr_t)p;
}

static __device__ inline unsigned int pack2_bf16(float a, float b) {
    union { __bf16 h[2]; unsigned int u; } p;
    p.h[0] = (__bf16)a;   // low 16 bits = even K
    p.h[1] = (__bf16)b;   // high 16 bits = odd K
    return p.u;
}

// ---------------------------------------------------------------------------
// Kernel 1: repack W_conv1 [HID=256, C=512] f32 -> bf16 WMMA A-fragments.
// Fragment (mt,kt) covers M = mt*16..+15, K = kt*32..+31.
// Per ISA 7.12.2 (16-bit A 16x32): lane<16 holds row M=lane, K-halves {0..7,16..23};
// lane>=16 holds row M=lane-16, K-halves {8..15,24..31}; 2 K per dword.
// Layout in ws: dword index = ((mt*16+kt)*32 + lane)*8 + j   (32B per lane -> b128 pairs)
// ---------------------------------------------------------------------------
__global__ __launch_bounds__(256) void pack_A_kernel(const float* __restrict__ W,
                                                     unsigned int* __restrict__ out) {
    int gid  = blockIdx.x * 256 + threadIdx.x;      // 8192 = 256 frags * 32 lanes
    int frag = gid >> 5;
    int lane = gid & 31;
    int mt = frag >> 4;
    int kt = frag & 15;
    int m    = mt * 16 + (lane & 15);
    int kadd = (lane >= 16) ? 8 : 0;

    unsigned int v[8];
#pragma unroll
    for (int j = 0; j < 8; ++j) {
        int k0 = kt * 32 + kadd + ((j & 4) ? 16 : 0) + 2 * (j & 3);
        v[j] = pack2_bf16(W[m * CDIM + k0], W[m * CDIM + k0 + 1]);
    }
    uint4* o = (uint4*)(out + (size_t)gid * 8);
    o[0] = make_uint4(v[0], v[1], v[2], v[3]);
    o[1] = make_uint4(v[4], v[5], v[6], v[7]);
}

// ---------------------------------------------------------------------------
// Kernel 2: hp[b][k] = h[b] . W_lin[k] + b_lin[k]      (tiny: 32x256)
// ---------------------------------------------------------------------------
__global__ __launch_bounds__(256) void hproj_kernel(const float* __restrict__ h,
                                                    const float* __restrict__ W_lin,
                                                    const float* __restrict__ b_lin,
                                                    float* __restrict__ hp) {
    int gid = blockIdx.x * 256 + threadIdx.x;       // 8192
    int b = gid >> 8;
    int k = gid & 255;
    const float4* hr = (const float4*)(h + (size_t)b * MEMD);
    const float4* wr = (const float4*)(W_lin + (size_t)k * MEMD);
    float acc = b_lin[k];
#pragma unroll 4
    for (int j = 0; j < MEMD / 4; ++j) {
        float4 a = hr[j], w = wr[j];
        acc += a.x * w.x + a.y * w.y + a.z * w.z + a.w * w.w;
    }
    hp[gid] = acc;
}

// ---------------------------------------------------------------------------
// Kernel 3 (main): per batch, X3 = W_conv1 * x[b]  via v_wmma_f32_16x16x32_bf16,
// fused tanh(+hp) and W_attn reduction -> a2[b][s].
// Grid: 256 blocks = 32 batches x 8 column-slabs(128). Block = 8 waves,
// wave w owns 16 columns.
//
// Data movement: the 32(K) x 128(col) f32 x-slab for each K-step is staged
// into LDS with double-buffered GLOBAL_LOAD_ASYNC_TO_LDS_B32 (128B-coalesced,
// no VGPR round trip, overlapped with compute via ASYNCcnt). LDS tile is
// column-major (stride 36 floats: 16B-aligned b128 reads, bank-spread), so
// each lane's B-fragment is 4x ds_load_b128.
// ---------------------------------------------------------------------------
#define TSTRIDE 36          // floats per column in LDS tile (padded from 32)

__global__ __launch_bounds__(256) void attn_gemm_kernel(const float* __restrict__ x,
                                                        const unsigned int* __restrict__ packedA,
                                                        const float* __restrict__ hp,
                                                        const float* __restrict__ wattn,
                                                        float* __restrict__ a2) {
    const int b      = blockIdx.x >> 3;
    const int colblk = blockIdx.x & 7;
    const int tid    = threadIdx.x;
    const int wave   = tid >> 5;
    const int lane   = tid & 31;

    __shared__ float s_tile[2][128 * TSTRIDE];      // 2 x 18KB x-slab buffers
    __shared__ float s_hp[HID];
    __shared__ float s_wa[HID];
    s_hp[tid] = hp[b * HID + tid];
    s_wa[tid] = wattn[tid];

    const int colbase = colblk * 128;
    const float* xb = x + (size_t)b * (CDIM * HW) + colbase;

    // Issue one 32x128 f32 slab (K-step kt) into LDS buffer bufi.
    // 4096 elements / 256 threads = 16 async b32 per thread; lanes cover
    // consecutive columns -> 128B-coalesced global transactions.
    auto issue_tile = [&](int kt, int bufi) {
        const float* base = xb + (size_t)(kt * 32) * HW;
#pragma unroll
        for (int i = 0; i < 16; ++i) {
            int e   = i * 256 + tid;
            int row = e >> 7;            // K within slab (0..31)
            int c   = e & 127;           // column within slab
            const float* g = base + (size_t)row * HW + c;
            unsigned int l = lds_off(&s_tile[bufi][c * TSTRIDE + row]);
            ASYNC_LOAD_B32(l, g);
        }
    };

    const int col16 = (lane & 15);                  // column within wave tile
    const int col   = colbase + wave * 16 + col16;  // global column
    const int khalf = (lane >> 4) * 8;              // B-frag K base for half-wave
    const int mofs  = (lane >> 4) * 8;              // C/D layout: hi lanes hold M+8

    v8f acc[16] = {};                               // 16 M-tiles of f32 accumulators

    issue_tile(0, 0);
    WAIT_ASYNCCNT0();
    __syncthreads();

    for (int kt = 0; kt < 16; ++kt) {
        const int cur = kt & 1;
        if (kt + 1 < 16) issue_tile(kt + 1, cur ^ 1);

        // ---- B fragment from LDS: K rows {khalf..+7, 16+khalf..+7} of this lane's column
        const float4* bp =
            (const float4*)&s_tile[cur][(wave * 16 + col16) * TSTRIDE + khalf];
        float4 f0 = bp[0], f1 = bp[1], f2 = bp[4], f3 = bp[5];
        v16bf bf;
        bf[0]  = (__bf16)f0.x;  bf[1]  = (__bf16)f0.y;
        bf[2]  = (__bf16)f0.z;  bf[3]  = (__bf16)f0.w;
        bf[4]  = (__bf16)f1.x;  bf[5]  = (__bf16)f1.y;
        bf[6]  = (__bf16)f1.z;  bf[7]  = (__bf16)f1.w;
        bf[8]  = (__bf16)f2.x;  bf[9]  = (__bf16)f2.y;
        bf[10] = (__bf16)f2.z;  bf[11] = (__bf16)f2.w;
        bf[12] = (__bf16)f3.x;  bf[13] = (__bf16)f3.y;
        bf[14] = (__bf16)f3.z;  bf[15] = (__bf16)f3.w;

        // ---- 16 M-tiles share this B fragment (A frags are L2-resident)
        const unsigned int* abase = packedA + ((size_t)kt * 32 + lane) * 8;
#pragma unroll
        for (int mt = 0; mt < 16; ++mt) {
            v16bf af = *(const v16bf*)(abase + (size_t)mt * (16 * 32 * 8));
            acc[mt] = __builtin_amdgcn_wmma_f32_16x16x32_bf16(
                false, af, false, bf, (short)0, acc[mt], false, false);
        }

        WAIT_ASYNCCNT0();       // next tile landed in LDS
        __syncthreads();        // all waves done reading cur, next buffer ready
    }

    // ---- fused epilogue: colsum = sum_m wattn[m] * tanh(x3[m][col] + hp[b][m]) ----
    float colsum = 0.f;
#pragma unroll
    for (int mt = 0; mt < 16; ++mt) {
#pragma unroll
        for (int i = 0; i < 8; ++i) {
            int m = mt * 16 + mofs + i;
            float v = tanhf(acc[mt][i] + s_hp[m]);
            colsum += v * s_wa[m];
        }
    }
    // lanes l and l+16 hold partials for the same column
    colsum += __shfl_xor(colsum, 16, 32);
    if (lane < 16) a2[(size_t)b * HW + col] = colsum;
}

// ---------------------------------------------------------------------------
// Kernel 4: row softmax over HW=1024 per batch. 32 blocks x 256 threads (x4 vals).
// ---------------------------------------------------------------------------
__global__ __launch_bounds__(256) void softmax_kernel(const float* __restrict__ a2,
                                                      float* __restrict__ a3) {
    const int b   = blockIdx.x;
    const int tid = threadIdx.x;
    __shared__ float red[256];

    float4 v = ((const float4*)(a2 + (size_t)b * HW))[tid];
    float m = fmaxf(fmaxf(v.x, v.y), fmaxf(v.z, v.w));
    red[tid] = m;
    __syncthreads();
    for (int s = 128; s > 0; s >>= 1) {
        if (tid < s) red[tid] = fmaxf(red[tid], red[tid + s]);
        __syncthreads();
    }
    const float rowmax = red[0];
    __syncthreads();

    float e0 = expf(v.x - rowmax), e1 = expf(v.y - rowmax);
    float e2 = expf(v.z - rowmax), e3 = expf(v.w - rowmax);
    red[tid] = e0 + e1 + e2 + e3;
    __syncthreads();
    for (int s = 128; s > 0; s >>= 1) {
        if (tid < s) red[tid] += red[tid + s];
        __syncthreads();
    }
    const float inv = 1.f / red[0];
    float4 o = make_float4(e0 * inv, e1 * inv, e2 * inv, e3 * inv);
    ((float4*)(a3 + (size_t)b * HW))[tid] = o;
}

// ---------------------------------------------------------------------------
// Kernel 5: context[b][c] = sum_s x[b][c][s] * a3[b][s]. One wave32 per (b,c).
// ---------------------------------------------------------------------------
__global__ __launch_bounds__(256) void context_kernel(const float* __restrict__ x,
                                                      const float* __restrict__ a3,
                                                      float* __restrict__ ctx) {
    const int gwave = (blockIdx.x * 256 + threadIdx.x) >> 5;   // 16384 = 32*512
    const int lane  = threadIdx.x & 31;
    const int b = gwave >> 9;
    const int c = gwave & 511;

    const float4* xr = (const float4*)(x + ((size_t)(b * CDIM + c) * HW));
    const float4* ar = (const float4*)(a3 + ((size_t)b * HW));

    float acc = 0.f;
#pragma unroll
    for (int i = 0; i < 8; ++i) {
        float4 xv = xr[i * 32 + lane];
        float4 av = ar[i * 32 + lane];
        acc += xv.x * av.x + xv.y * av.y + xv.z * av.z + xv.w * av.w;
    }
#pragma unroll
    for (int off = 16; off > 0; off >>= 1) acc += __shfl_down(acc, off, 32);
    if (lane == 0) ctx[gwave] = acc;
}

// ---------------------------------------------------------------------------
extern "C" void kernel_launch(void* const* d_in, const int* in_sizes, int n_in,
                              void* d_out, int out_size, void* d_ws, size_t ws_size,
                              hipStream_t stream) {
    (void)in_sizes; (void)n_in; (void)out_size; (void)ws_size;

    const float* x       = (const float*)d_in[0];   // [32,512,32,32]
    const float* h       = (const float*)d_in[1];   // [32,256]
    const float* W_conv1 = (const float*)d_in[2];   // [256,512]
    const float* W_lin   = (const float*)d_in[3];   // [256,256]
    const float* b_lin   = (const float*)d_in[4];   // [256]
    const float* W_attn  = (const float*)d_in[5];   // [256]

    float* a3_out  = (float*)d_out;                 // [32,1024]
    float* ctx_out = a3_out + BATCH * HW;           // [32,512]

    unsigned char* ws = (unsigned char*)d_ws;
    unsigned int* packedA = (unsigned int*)ws;                       // 262144 B
    float*        hp      = (float*)(ws + 262144);                   //  32768 B
    float*        a2      = (float*)(ws + 262144 + 32768);           // 131072 B

    pack_A_kernel  <<<32,   256, 0, stream>>>(W_conv1, packedA);
    hproj_kernel   <<<32,   256, 0, stream>>>(h, W_lin, b_lin, hp);
    attn_gemm_kernel<<<256, 256, 0, stream>>>(x, packedA, hp, W_attn, a2);
    softmax_kernel <<<32,   256, 0, stream>>>(a2, a3_out);
    context_kernel <<<2048, 256, 0, stream>>>(x, a3_out, ctx_out);
}
